// NerfHash_89790586290725
// MI455X (gfx1250) — compile-verified, matrix operand order checked by reading, not run on hardware
//
#include <hip/hip_runtime.h>

typedef __attribute__((ext_vector_type(16))) _Float16 v16h;
typedef __attribute__((ext_vector_type(8)))  float    v8f;

#define TBITS   19
#define TMASK   ((1u << TBITS) - 1u)
#define WPB     8        // waves per block (256 threads)
#define TPW     4        // tiles (of 16 points) per wave
#define HT_LD   96       // activation staging row stride (halves)

// ---- bias offsets (floats) ----
#define B0 0
#define B1 64
#define B2 128
#define B3 192          // fb3: 65 used (+3 pad)
#define R0 260
#define R1 324
#define R2 388          // rb2: 3 used (+1 pad)
#define BIASF 392

// ---- f16 region offsets (halves) ----
#define W0T 0           // [n=64][k=64]  (k<48 real)
#define W1T 4096
#define W2T 8192
#define W3T 12288       // fw3 cols 1..64 (feat part)
#define R0T 16384       // [n=64][k=96]  (k<73 real)
#define R1T 22528
#define R2T 26624       // [n=16][k=64]  (n<3 real)
#define W3D 27648       // fw3 col 0 (density), 64 halves
#define HTB 27712       // per-wave activation tiles
#define SMEM_HALVES (HTB + WPB * 16 * HT_LD)
#define SMEM_BYTES  (BIASF * 4 + SMEM_HALVES * 2)

__device__ __forceinline__ float gelu_f(float x) {
  return 0.5f * x * (1.0f + erff(x * 0.70710678118654752440f));
}
__device__ __forceinline__ float softplus_f(float x) {
  return (x > 20.0f) ? x : log1pf(expf(x));
}
__device__ __forceinline__ float sigmoid_f(float x) {
  return 1.0f / (1.0f + expf(-x));
}

// RES[l] = floor(16 * ((2048/16)^(1/23))^l) = floor(2^(4 + 7l/23))
__device__ __forceinline__ float res_of(int lo, int hh) {
  constexpr float RESV[24] = {
      16.f,  19.f,  24.f,  30.f,  37.f,  45.f,  56.f,  70.f,
      86.f, 106.f, 131.f, 162.f, 201.f, 248.f, 306.f, 378.f,
     467.f, 577.f, 713.f, 880.f, 1087.f, 1343.f, 1658.f, 2047.f};
  return hh ? RESV[lo + 4] : RESV[lo];   // lo is compile-time in unrolled loops
}

__device__ __forceinline__ v8f wmma_f16(v16h a, v16h b, v8f c) {
  return __builtin_amdgcn_wmma_f32_16x16x32_f16(
      /*neg_a=*/false, a, /*neg_b=*/false, b,
      /*c_mod=*/(short)0, c, /*reuse_a=*/false, /*reuse_b=*/false);
}

// Load one 16x32 f16 fragment (A or B^T) from an LDS row in the ISA K-pattern:
// VGPR r holds K pair at k = 32*kstep + (r<4?0:16) + 8*half + 2*(r%4).
__device__ __forceinline__ v16h frag_load(const _Float16* rp, int kstep, int hh) {
  v16h f;
  const int kb = kstep * 32 + hh * 8;
#pragma unroll
  for (int r = 0; r < 8; ++r) {
    const int k = kb + ((r & 4) << 2) + ((r & 3) << 1);
    f[2 * r]     = rp[k];
    f[2 * r + 1] = rp[k + 1];
  }
  return f;
}

// One hash-grid level: trilinear interp of 8 corners from the table.
__device__ __forceinline__ void enc_level(int lev, float res,
                                          float x0, float y0, float z0,
                                          const float* __restrict__ tbl,
                                          float& f0, float& f1) {
  const float fx = x0 * res, fy = y0 * res, fz = z0 * res;
  const float gx = floorf(fx), gy = floorf(fy), gz = floorf(fz);
  const float wx = fx - gx, wy = fy - gy, wz = fz - gz;
  const unsigned cx = (unsigned)gx, cy = (unsigned)gy, cz = (unsigned)gz;
  const float* lt = tbl + (((size_t)lev) << (TBITS + 1));
  f0 = 0.f; f1 = 0.f;
#pragma unroll
  for (int cc = 0; cc < 8; ++cc) {
    const unsigned bx = (cc >> 2) & 1u, by = (cc >> 1) & 1u, bz = cc & 1u;
    const unsigned idx =
        ((cx + bx) * 1u ^ (cy + by) * 2654435761u ^ (cz + bz) * 805459861u) & TMASK;
    const float2 fv = *reinterpret_cast<const float2*>(lt + ((size_t)idx << 1));
    const float w = (bx ? wx : 1.f - wx) * (by ? wy : 1.f - wy) * (bz ? wz : 1.f - wz);
    f0 += fv.x * w; f1 += fv.y * w;
  }
}

// D(16x64) = gelu(A(16xK) @ Wt^T + bias); Wt is [n][ldw] f16 in LDS; result -> ht.
template <int NS>
__device__ __forceinline__ void layer_fwd(const v16h* a, const _Float16* wt, int ldw,
                                          const float* bias, _Float16* ht,
                                          int lane, int hh) {
  const int col = lane & 15;
#pragma unroll
  for (int nt = 0; nt < 4; ++nt) {
    const int n = nt * 16 + col;
    const _Float16* rp = wt + n * ldw;
    v8f c = {0.f, 0.f, 0.f, 0.f, 0.f, 0.f, 0.f, 0.f};
#pragma unroll
    for (int s = 0; s < NS; ++s)
      c = wmma_f16(a[s], frag_load(rp, s, hh), c);
    const float bv = bias[n];
#pragma unroll
    for (int r = 0; r < 8; ++r)   // D layout: row = r + 8*(lane>=16), col = lane%16
      ht[(r + 8 * hh) * HT_LD + n] = (_Float16)gelu_f(c[r] + bv);
  }
}

__global__ __launch_bounds__(WPB * 32) void nerf_fused(
    const float* __restrict__ pts, const float* __restrict__ dirs,
    const float* __restrict__ tbl,
    const float* __restrict__ fw0, const float* __restrict__ fb0,
    const float* __restrict__ fw1, const float* __restrict__ fb1,
    const float* __restrict__ fw2, const float* __restrict__ fb2,
    const float* __restrict__ fw3, const float* __restrict__ fb3,
    const float* __restrict__ rw0, const float* __restrict__ rb0,
    const float* __restrict__ rw1, const float* __restrict__ rb1,
    const float* __restrict__ rw2, const float* __restrict__ rb2,
    float* __restrict__ out, int Np, int nTiles) {
  extern __shared__ char smem[];
  float*    bias = reinterpret_cast<float*>(smem);
  _Float16* hb   = reinterpret_cast<_Float16*>(smem + BIASF * 4);

  const int tid = threadIdx.x, nth = blockDim.x;

  // ---- stage all weights (transposed, f16, zero-padded) + biases into LDS ----
  for (int i = tid; i < 64 * 64; i += nth) {
    const int n = i >> 6, k = i & 63;
    hb[W0T + i] = (_Float16)((k < 48) ? fw0[k * 64 + n] : 0.f);
    hb[W1T + i] = (_Float16)fw1[k * 64 + n];
    hb[W2T + i] = (_Float16)fw2[k * 64 + n];
    hb[W3T + i] = (_Float16)fw3[k * 65 + 1 + n];   // feat part (cols 1..64)
    hb[R1T + i] = (_Float16)rw1[k * 64 + n];
  }
  for (int i = tid; i < 64 * 96; i += nth) {
    const int n = i / 96, k = i - n * 96;
    hb[R0T + i] = (_Float16)((k < 73) ? rw0[k * 64 + n] : 0.f);
  }
  for (int i = tid; i < 16 * 64; i += nth) {
    const int n = i >> 6, k = i & 63;
    hb[R2T + i] = (_Float16)((n < 3) ? rw2[k * 3 + n] : 0.f);
  }
  for (int i = tid; i < 64; i += nth) {
    hb[W3D + i] = (_Float16)fw3[i * 65];           // density column
    bias[B0 + i] = fb0[i]; bias[B1 + i] = fb1[i]; bias[B2 + i] = fb2[i];
    bias[R0 + i] = rb0[i]; bias[R1 + i] = rb1[i];
  }
  for (int i = tid; i < 68; i += nth) bias[B3 + i] = (i < 65) ? fb3[i] : 0.f;
  if (tid < 4) bias[R2 + tid] = (tid < 3) ? rb2[tid] : 0.f;
  __syncthreads();

  const int wave = tid >> 5, lane = tid & 31;
  const int hh = lane >> 4, mrow = lane & 15;
  _Float16* ht = hb + HTB + wave * 16 * HT_LD;
  const _Float16* myrow = ht + mrow * HT_LD;

  // zero once: pad cols (>=73) must stay 0 across all tiles
  for (int i = lane; i < 16 * HT_LD; i += 32) ht[i] = (_Float16)0.f;

  const long tile0 = ((long)blockIdx.x * WPB + wave) * TPW;

  for (int t = 0; t < TPW; ++t) {
    const long tile = tile0 + t;
    if (tile >= nTiles) break;                   // wave-uniform
    const long base = tile * 16;
    long pt = base + mrow; if (pt >= Np) pt = Np - 1;

    // ---- SH l<=2 direction encoding -> staging cols 64..72 ----
    if (hh == 0) {
      const float dx = dirs[pt * 3 + 0], dy = dirs[pt * 3 + 1], dz = dirs[pt * 3 + 2];
      _Float16* s = ht + mrow * HT_LD + 64;
      s[0] = (_Float16)0.28209479177387814f;
      s[1] = (_Float16)(-0.48860251190291987f * dy);
      s[2] = (_Float16)( 0.48860251190291987f * dz);
      s[3] = (_Float16)(-0.48860251190291987f * dx);
      s[4] = (_Float16)( 1.0925484305920792f * dx * dy);
      s[5] = (_Float16)(-1.0925484305920792f * dy * dz);
      s[6] = (_Float16)( 0.31539156525252005f * (3.f * dz * dz - 1.f));
      s[7] = (_Float16)(-1.0925484305920792f * dx * dz);
      s[8] = (_Float16)( 0.5462742152960396f * (dx * dx - dy * dy));
    }

    // ---- hash encode directly into WMMA A-fragments (K pad 48->64) ----
    const float px = pts[pt * 3 + 0], py = pts[pt * 3 + 1], pz = pts[pt * 3 + 2];
    const float x0 = (px + 1.f) * 0.5f, y0 = (py + 1.f) * 0.5f, z0 = (pz + 1.f) * 0.5f;

    v16h a[3];
#pragma unroll
    for (int r = 0; r < 8; ++r) {          // step 0: k 0..31 -> levels {4h..4h+3, 8+4h..11+4h}
      const int lo = ((r & 4) ? 8 : 0) + (r & 3);
      const int lev = lo + 4 * hh;
      float f0, f1; enc_level(lev, res_of(lo, hh), x0, y0, z0, tbl, f0, f1);
      a[0][2 * r] = (_Float16)f0; a[0][2 * r + 1] = (_Float16)f1;
    }
#pragma unroll
    for (int r = 0; r < 4; ++r) {          // step 1 low: k 32..47 -> levels 16+4h..19+4h
      const int lo = 16 + r;
      const int lev = lo + 4 * hh;
      float f0, f1; enc_level(lev, res_of(lo, hh), x0, y0, z0, tbl, f0, f1);
      a[1][2 * r] = (_Float16)f0; a[1][2 * r + 1] = (_Float16)f1;
    }
#pragma unroll
    for (int r = 4; r < 8; ++r) {          // step 1 high: K pad
      a[1][2 * r] = (_Float16)0.f; a[1][2 * r + 1] = (_Float16)0.f;
    }

    // ---- feature MLP ----
    layer_fwd<2>(a, hb + W0T, 64, bias + B0, ht, lane, hh);

    a[0] = frag_load(myrow, 0, hh); a[1] = frag_load(myrow, 1, hh);
    layer_fwd<2>(a, hb + W1T, 64, bias + B1, ht, lane, hh);

    a[0] = frag_load(myrow, 0, hh); a[1] = frag_load(myrow, 1, hh);
    layer_fwd<2>(a, hb + W2T, 64, bias + B2, ht, lane, hh);

    // density (fw3 col 0) from h2 before layer3 overwrites staging
    a[0] = frag_load(myrow, 0, hh); a[1] = frag_load(myrow, 1, hh);
    if (hh == 0) {
      float acc = bias[B3 + 0];
#pragma unroll
      for (int k = 0; k < 64; ++k)
        acc += (float)ht[mrow * HT_LD + k] * (float)hb[W3D + k];
      if (base + mrow < Np) out[(size_t)3 * Np + base + mrow] = softplus_f(acc);
    }
    layer_fwd<2>(a, hb + W3T, 64, bias + B3 + 1, ht, lane, hh);   // feat_rgb -> cols 0..63

    // ---- rgb head: K = 64 feat + 9 SH (+pad) = 3 K-steps ----
    a[0] = frag_load(myrow, 0, hh); a[1] = frag_load(myrow, 1, hh);
    a[2] = frag_load(myrow, 2, hh);
    layer_fwd<3>(a, hb + R0T, 96, bias + R0, ht, lane, hh);

    a[0] = frag_load(myrow, 0, hh); a[1] = frag_load(myrow, 1, hh);
    layer_fwd<2>(a, hb + R1T, 64, bias + R1, ht, lane, hh);

    // final 64->3 (one N-tile) + sigmoid -> global
    a[0] = frag_load(myrow, 0, hh); a[1] = frag_load(myrow, 1, hh);
    {
      const int n = lane & 15;
      const _Float16* rp = hb + R2T + n * 64;
      v8f c = {0.f, 0.f, 0.f, 0.f, 0.f, 0.f, 0.f, 0.f};
      c = wmma_f16(a[0], frag_load(rp, 0, hh), c);
      c = wmma_f16(a[1], frag_load(rp, 1, hh), c);
      if (n < 3) {
        const float bv = bias[R2 + n];
#pragma unroll
        for (int r = 0; r < 8; ++r) {
          const long pr = base + r + 8 * hh;
          if (pr < Np) out[pr * 3 + n] = sigmoid_f(c[r] + bv);
        }
      }
    }
  }
}

extern "C" void kernel_launch(void* const* d_in, const int* in_sizes, int n_in,
                              void* d_out, int out_size, void* d_ws, size_t ws_size,
                              hipStream_t stream) {
  const float* pts = (const float*)d_in[0];
  const float* dirs = (const float*)d_in[1];
  const float* tbl  = (const float*)d_in[2];
  const float* fw0 = (const float*)d_in[3];  const float* fb0 = (const float*)d_in[4];
  const float* fw1 = (const float*)d_in[5];  const float* fb1 = (const float*)d_in[6];
  const float* fw2 = (const float*)d_in[7];  const float* fb2 = (const float*)d_in[8];
  const float* fw3 = (const float*)d_in[9];  const float* fb3 = (const float*)d_in[10];
  const float* rw0 = (const float*)d_in[11]; const float* rb0 = (const float*)d_in[12];
  const float* rw1 = (const float*)d_in[13]; const float* rb1 = (const float*)d_in[14];
  const float* rw2 = (const float*)d_in[15]; const float* rb2 = (const float*)d_in[16];
  float* out = (float*)d_out;

  const int Np = in_sizes[0] / 3;
  const int nTiles = (Np + 15) / 16;
  const int tilesPerBlock = WPB * TPW;
  const int blocks = (nTiles + tilesPerBlock - 1) / tilesPerBlock;

  // dynamic LDS is ~81.6 KB (> 64 KB default cap): raise the limit.
  (void)hipFuncSetAttribute(reinterpret_cast<const void*>(&nerf_fused),
                            hipFuncAttributeMaxDynamicSharedMemorySize,
                            SMEM_BYTES);

  nerf_fused<<<dim3(blocks), dim3(WPB * 32), SMEM_BYTES, stream>>>(
      pts, dirs, tbl, fw0, fb0, fw1, fb1, fw2, fb2, fw3, fb3,
      rw0, rb0, rw1, rb1, rw2, rb2, out, Np, nTiles);
}
